// EdgeNetEmbed_47588237639715
// MI455X (gfx1250) — compile-verified, matrix operand order checked by reading, not run on hardware
//
#include <hip/hip_runtime.h>
#include <hip/hip_bf16.h>

typedef _Float16 h16;
typedef __attribute__((ext_vector_type(16))) _Float16 v16h;
typedef __attribute__((ext_vector_type(8)))  _Float16 v8h;
typedef __attribute__((ext_vector_type(8)))  float    v8f;
typedef __attribute__((ext_vector_type(2)))  float    v2f;

#define WPB 4            // waves per block (wave32)
#define TPB (WPB * 32)

__device__ __forceinline__ v8f wmma16(v16h a, v16h b, v8f c) {
  return __builtin_amdgcn_wmma_f32_16x16x32_f16(false, a, false, b, (short)0, c,
                                                false, false);
}
__device__ __forceinline__ v8f wmma4(v2f a, v2f b, v8f c) {
  return __builtin_amdgcn_wmma_f32_16x16x4_f32(false, a, false, b, (short)0, c,
                                               false, false);
}

// A-operand fragment of W^T for 16x16x32 f16 WMMA.
// W is [K=32][nout] row-major f32 in global memory; m = output-feature row.
// Lane L (<16) holds row m with K {0..7,16..23}; lane L+16 holds K {8..15,24..31}.
__device__ __forceinline__ v16h wfragT(const float* __restrict__ W, int nout,
                                       int m, int lane) {
  int kb = (lane < 16) ? 0 : 8;
  v16h a;
#pragma unroll
  for (int i = 0; i < 8; ++i) {
    a[i]     = (h16)W[(kb + i) * nout + m];
    a[8 + i] = (h16)W[(kb + 16 + i) * nout + m];
  }
  return a;
}

// A-operand fragment of W^T for 16x16x4 f32 WMMA (K = 4).
__device__ __forceinline__ v2f wfrag4(const float* __restrict__ W, int nout,
                                      int m, int lane) {
  int k0 = (lane < 16) ? 0 : 2;
  v2f a;
  a[0] = W[k0 * nout + m];
  a[1] = W[(k0 + 1) * nout + m];
  return a;
}

// Per-row bias accumulator: C[i] = bias[M], M = mtile*16 + rb + i.
__device__ __forceinline__ v8f biasrows(const float* __restrict__ bias, int mtile,
                                        int lane) {
  int mb = mtile * 16 + ((lane < 16) ? 0 : 8);
  v8f c;
#pragma unroll
  for (int i = 0; i < 8; ++i) c[i] = bias[mb + i];
  return c;
}

// f32 -> f16 packed convert, then ReLU in the f16 domain (monotone, so
// cvt-then-max == max-then-cvt for ReLU; avoids the f32 canonicalize+max pair).
__device__ __forceinline__ v8h relu8h(v8f d) {
  v8h r;
#pragma unroll
  for (int i = 0; i < 8; ++i) r[i] = (h16)d[i];
#pragma unroll
  for (int i = 0; i < 8; ++i) r[i] = (r[i] > (h16)0.f) ? r[i] : (h16)0.f;
  return r;
}

__device__ __forceinline__ v16h combine16(v8h lo, v8h hi) {
  v16h b;
#pragma unroll
  for (int i = 0; i < 8; ++i) {
    b[i]     = lo[i];
    b[8 + i] = hi[i];
  }
  return b;
}

__device__ __forceinline__ v8f zero8() {
  v8f z;
#pragma unroll
  for (int i = 0; i < 8; ++i) z[i] = 0.f;
  return z;
}

// ---------------------------------------------------------------- BN stats
__global__ void k_bnstats(const float* __restrict__ x, int N, float* __restrict__ bnws) {
  float s[4] = {0, 0, 0, 0}, q[4] = {0, 0, 0, 0};
  for (int n = blockIdx.x * blockDim.x + threadIdx.x; n < N;
       n += gridDim.x * blockDim.x) {
#pragma unroll
    for (int f = 0; f < 4; ++f) {
      float v = x[n * 4 + f];
      s[f] += v;
      q[f] += v * v;
    }
  }
#pragma unroll
  for (int off = 16; off > 0; off >>= 1) {
#pragma unroll
    for (int f = 0; f < 4; ++f) {
      s[f] += __shfl_down(s[f], off);
      q[f] += __shfl_down(q[f], off);
    }
  }
  if ((threadIdx.x & 31) == 0) {
#pragma unroll
    for (int f = 0; f < 4; ++f) {
      atomicAdd(&bnws[f], s[f]);
      atomicAdd(&bnws[4 + f], q[f]);
    }
  }
}

// ---------------------------------------------------------------- degree counts
__global__ void k_count(const int* __restrict__ ei, int E, float* __restrict__ cnt) {
  for (int e = blockIdx.x * blockDim.x + threadIdx.x; e < E;
       e += gridDim.x * blockDim.x)
    atomicAdd(&cnt[ei[E + e]], 1.0f);
}

// ---------------------------------------------------------------- embed: BN + 4->32->32->32
__global__ void k_embed(const float* __restrict__ x, int N,
                        const float* __restrict__ bnws,
                        const float* __restrict__ bn_g, const float* __restrict__ bn_b,
                        const float* __restrict__ w0, const float* __restrict__ b0,
                        const float* __restrict__ w1, const float* __restrict__ b1,
                        const float* __restrict__ w2, const float* __restrict__ b2,
                        h16* __restrict__ hout) {
  int lane = threadIdx.x & 31, wave = threadIdx.x >> 5;
  int m0 = lane & 15, m1 = m0 + 16;
  int rb = (lane < 16) ? 0 : 8;

  v2f  wa0a = wfrag4(w0, 32, m0, lane), wa0b = wfrag4(w0, 32, m1, lane);
  v16h wa1a = wfragT(w1, 32, m0, lane), wa1b = wfragT(w1, 32, m1, lane);
  v16h wa2a = wfragT(w2, 32, m0, lane), wa2b = wfragT(w2, 32, m1, lane);
  v8f b0a = biasrows(b0, 0, lane), b0b = biasrows(b0, 1, lane);
  v8f b1a = biasrows(b1, 0, lane), b1b = biasrows(b1, 1, lane);
  v8f b2a = biasrows(b2, 0, lane), b2b = biasrows(b2, 1, lane);

  // BN affine for this lane's two K features
  float invN = 1.0f / (float)N;
  int   k0f  = (lane < 16) ? 0 : 2;
  float scl[2], sft[2];
#pragma unroll
  for (int j = 0; j < 2; ++j) {
    int   f   = k0f + j;
    float mu  = bnws[f] * invN;
    float var = bnws[4 + f] * invN - mu * mu;
    float sc  = bn_g[f] * rsqrtf(var + 1e-5f);
    scl[j] = sc;
    sft[j] = bn_b[f] - mu * sc;
  }

  int ntiles = (N + 15) >> 4;
  int wid = blockIdx.x * WPB + wave, nw = gridDim.x * WPB;
  for (int tile = wid; tile < ntiles; tile += nw) {
    int te = tile << 4;
    int node = te + (lane & 15);
    bool valid = node < N;
    if (!valid) node = N - 1;

    v2f xv = *(const v2f*)(x + (size_t)node * 4 + k0f);
    v2f bx;
    bx[0] = xv[0] * scl[0] + sft[0];
    bx[1] = xv[1] * scl[1] + sft[1];

    v8f d0 = wmma4(wa0a, bx, b0a);
    v8f d1 = wmma4(wa0b, bx, b0b);
    v16h bf = combine16(relu8h(d0), relu8h(d1));
    d0 = wmma16(wa1a, bf, b1a);
    d1 = wmma16(wa1b, bf, b1b);
    bf = combine16(relu8h(d0), relu8h(d1));
    d0 = wmma16(wa2a, bf, b2a);
    d1 = wmma16(wa2b, bf, b2b);

    if (valid) {
      *(v8h*)(hout + (size_t)node * 32 + rb)      = relu8h(d0);
      *(v8h*)(hout + (size_t)node * 32 + 16 + rb) = relu8h(d1);
    }
  }
}

// ---------------------------------------------------------------- EdgeConv1: 64->32->32->2 + atomic sum
// Two independent edge tiles per wave iteration: the two WMMA chains interleave,
// filling the WMMA->VALU hazard slots and hiding gather latency.
__global__ void k_conv1(const int* __restrict__ ei, int E,
                        const h16* __restrict__ hbuf,
                        const float* __restrict__ w0, const float* __restrict__ b0,
                        const float* __restrict__ w1, const float* __restrict__ b1,
                        const float* __restrict__ w2, const float* __restrict__ b2,
                        float* __restrict__ seg2) {
  int lane = threadIdx.x & 31, wave = threadIdx.x >> 5;
  int m0 = lane & 15, m1 = m0 + 16;
  int kb = (lane < 16) ? 0 : 8;

  v16h wk0a = wfragT(w0, 32, m0, lane);            // K rows 0..31  (x_i)
  v16h wk0b = wfragT(w0, 32, m1, lane);
  v16h wk1a = wfragT(w0 + 32 * 32, 32, m0, lane);  // K rows 32..63 (x_j - x_i)
  v16h wk1b = wfragT(w0 + 32 * 32, 32, m1, lane);
  v16h w1a = wfragT(w1, 32, m0, lane), w1b = wfragT(w1, 32, m1, lane);
  v16h w2f = wfragT(w2, 2, m0 & 1, lane);          // 32->2, rows 0..1 meaningful
  v8f b0a = biasrows(b0, 0, lane), b0b = biasrows(b0, 1, lane);
  v8f b1a = biasrows(b1, 0, lane), b1b = biasrows(b1, 1, lane);
  float b2v0 = b2[0], b2v1 = b2[1];
  v8f zc = zero8();

  int ntiles = (E + 15) >> 4;
  int npairs = (ntiles + 1) >> 1;
  int wid = blockIdx.x * WPB + wave, nw = gridDim.x * WPB;
  for (int p = wid; p < npairs; p += nw) {
    int teA = (2 * p) << 4, teB = teA + 16;
    int eA = teA + (lane & 15), eB = teB + (lane & 15);
    bool vA = eA < E, vB = eB < E;
    if (!vA) eA = E - 1;
    if (!vB) eB = E - 1;
    int sA = ei[eA], dA = ei[E + eA];
    int sB = ei[eB], dB = ei[E + eB];

    const h16* hdA = hbuf + (size_t)dA * 32;
    const h16* hsA = hbuf + (size_t)sA * 32;
    const h16* hdB = hbuf + (size_t)dB * 32;
    const h16* hsB = hbuf + (size_t)sB * 32;
    v8h xiaA = *(const v8h*)(hdA + kb), xibA = *(const v8h*)(hdA + kb + 16);
    v8h xjaA = *(const v8h*)(hsA + kb), xjbA = *(const v8h*)(hsA + kb + 16);
    v8h xiaB = *(const v8h*)(hdB + kb), xibB = *(const v8h*)(hdB + kb + 16);
    v8h xjaB = *(const v8h*)(hsB + kb), xjbB = *(const v8h*)(hsB + kb + 16);
    v16h bx0A = combine16(xiaA, xibA);
    v16h bx1A = combine16(xjaA - xiaA, xjbA - xibA);
    v16h bx0B = combine16(xiaB, xibB);
    v16h bx1B = combine16(xjaB - xiaB, xjbB - xibB);

    v8f d0A = b0a, d1A = b0b, d0B = b0a, d1B = b0b;
    d0A = wmma16(wk0a, bx0A, d0A);
    d0B = wmma16(wk0a, bx0B, d0B);
    d1A = wmma16(wk0b, bx0A, d1A);
    d1B = wmma16(wk0b, bx0B, d1B);
    d0A = wmma16(wk1a, bx1A, d0A);
    d0B = wmma16(wk1a, bx1B, d0B);
    d1A = wmma16(wk1b, bx1A, d1A);
    d1B = wmma16(wk1b, bx1B, d1B);

    v16h bfA = combine16(relu8h(d0A), relu8h(d1A));
    v16h bfB = combine16(relu8h(d0B), relu8h(d1B));
    d0A = wmma16(w1a, bfA, b1a);
    d0B = wmma16(w1a, bfB, b1a);
    d1A = wmma16(w1b, bfA, b1b);
    d1B = wmma16(w1b, bfB, b1b);

    bfA = combine16(relu8h(d0A), relu8h(d1A));
    bfB = combine16(relu8h(d0B), relu8h(d1B));
    v8f dzA = wmma16(w2f, bfA, zc);   // rows 0..1 = the two outputs
    v8f dzB = wmma16(w2f, bfB, zc);

    if (vA && lane < 16) {
      atomicAdd(&seg2[(size_t)dA * 2 + 0], fmaxf(dzA[0] + b2v0, 0.f));
      atomicAdd(&seg2[(size_t)dA * 2 + 1], fmaxf(dzA[1] + b2v1, 0.f));
    }
    if (vB && lane < 16) {
      atomicAdd(&seg2[(size_t)dB * 2 + 0], fmaxf(dzB[0] + b2v0, 0.f));
      atomicAdd(&seg2[(size_t)dB * 2 + 1], fmaxf(dzB[1] + b2v1, 0.f));
    }
  }
}

// ---------------------------------------------------------------- mean for hid[N,2]
__global__ void k_mean2(const float* __restrict__ seg2, const float* __restrict__ cnt,
                        int N, float* __restrict__ hid) {
  int total = N * 2;
  for (int i = blockIdx.x * blockDim.x + threadIdx.x; i < total;
       i += gridDim.x * blockDim.x) {
    float c = cnt[i >> 1];
    c = c > 1.f ? c : 1.f;
    hid[i] = seg2[i] / c;
  }
}

// ---------------------------------------------------------------- EdgeConv2: 4->32->32->32 + atomic sum
__global__ void k_conv2(const int* __restrict__ ei, int E,
                        const float* __restrict__ hid,
                        const float* __restrict__ w0, const float* __restrict__ b0,
                        const float* __restrict__ w1, const float* __restrict__ b1,
                        const float* __restrict__ w2, const float* __restrict__ b2,
                        float* __restrict__ seg32) {
  int lane = threadIdx.x & 31, wave = threadIdx.x >> 5;
  int m0 = lane & 15, m1 = m0 + 16;
  int rb = (lane < 16) ? 0 : 8;

  v2f  wa0a = wfrag4(w0, 32, m0, lane), wa0b = wfrag4(w0, 32, m1, lane);
  v16h wa1a = wfragT(w1, 32, m0, lane), wa1b = wfragT(w1, 32, m1, lane);
  v16h wa2a = wfragT(w2, 32, m0, lane), wa2b = wfragT(w2, 32, m1, lane);
  v8f b0a = biasrows(b0, 0, lane), b0b = biasrows(b0, 1, lane);
  v8f b1a = biasrows(b1, 0, lane), b1b = biasrows(b1, 1, lane);
  v8f b2a = biasrows(b2, 0, lane), b2b = biasrows(b2, 1, lane);

  int ntiles = (E + 15) >> 4;
  int npairs = (ntiles + 1) >> 1;
  int wid = blockIdx.x * WPB + wave, nw = gridDim.x * WPB;
  for (int p = wid; p < npairs; p += nw) {
    int teA = (2 * p) << 4, teB = teA + 16;
    int eA = teA + (lane & 15), eB = teB + (lane & 15);
    bool vA = eA < E, vB = eB < E;
    if (!vA) eA = E - 1;
    if (!vB) eB = E - 1;
    int sA = ei[eA], dA = ei[E + eA];
    int sB = ei[eB], dB = ei[E + eB];

    v2f xivA = *(const v2f*)(hid + (size_t)dA * 2);
    v2f xjvA = *(const v2f*)(hid + (size_t)sA * 2);
    v2f xivB = *(const v2f*)(hid + (size_t)dB * 2);
    v2f xjvB = *(const v2f*)(hid + (size_t)sB * 2);
    v2f bxA = (lane < 16) ? xivA : (xjvA - xivA);
    v2f bxB = (lane < 16) ? xivB : (xjvB - xivB);

    v8f d0A = wmma4(wa0a, bxA, b0a);
    v8f d0B = wmma4(wa0a, bxB, b0a);
    v8f d1A = wmma4(wa0b, bxA, b0b);
    v8f d1B = wmma4(wa0b, bxB, b0b);

    v16h bfA = combine16(relu8h(d0A), relu8h(d1A));
    v16h bfB = combine16(relu8h(d0B), relu8h(d1B));
    d0A = wmma16(wa1a, bfA, b1a);
    d0B = wmma16(wa1a, bfB, b1a);
    d1A = wmma16(wa1b, bfA, b1b);
    d1B = wmma16(wa1b, bfB, b1b);

    bfA = combine16(relu8h(d0A), relu8h(d1A));
    bfB = combine16(relu8h(d0B), relu8h(d1B));
    d0A = wmma16(wa2a, bfA, b2a);
    d0B = wmma16(wa2a, bfB, b2a);
    d1A = wmma16(wa2b, bfA, b2b);
    d1B = wmma16(wa2b, bfB, b2b);

    if (vA) {
      float* dst = seg32 + (size_t)dA * 32;
#pragma unroll
      for (int i = 0; i < 8; ++i) {
        atomicAdd(dst + rb + i, fmaxf(d0A[i], 0.f));
        atomicAdd(dst + 16 + rb + i, fmaxf(d1A[i], 0.f));
      }
    }
    if (vB) {
      float* dst = seg32 + (size_t)dB * 32;
#pragma unroll
      for (int i = 0; i < 8; ++i) {
        atomicAdd(dst + rb + i, fmaxf(d0B[i], 0.f));
        atomicAdd(dst + 16 + rb + i, fmaxf(d1B[i], 0.f));
      }
    }
  }
}

// ---------------------------------------------------------------- deembed: mean + 32->32->32->4
__global__ void k_deembed(const float* __restrict__ seg32, const float* __restrict__ cnt,
                          int N,
                          const float* __restrict__ w0, const float* __restrict__ b0,
                          const float* __restrict__ w1, const float* __restrict__ b1,
                          const float* __restrict__ w2, const float* __restrict__ b2,
                          float* __restrict__ out) {
  int lane = threadIdx.x & 31, wave = threadIdx.x >> 5;
  int m0 = lane & 15, m1 = m0 + 16;
  int kb = (lane < 16) ? 0 : 8;

  v16h wa0a = wfragT(w0, 32, m0, lane), wa0b = wfragT(w0, 32, m1, lane);
  v16h wa1a = wfragT(w1, 32, m0, lane), wa1b = wfragT(w1, 32, m1, lane);
  v16h wa2f = wfragT(w2, 4, m0 & 3, lane);   // 32->4, rows 0..3 meaningful
  v8f b0a = biasrows(b0, 0, lane), b0b = biasrows(b0, 1, lane);
  v8f b1a = biasrows(b1, 0, lane), b1b = biasrows(b1, 1, lane);
  float b2v[4] = {b2[0], b2[1], b2[2], b2[3]};
  v8f zc = zero8();

  int ntiles = (N + 15) >> 4;
  int wid = blockIdx.x * WPB + wave, nw = gridDim.x * WPB;
  for (int tile = wid; tile < ntiles; tile += nw) {
    int te = tile << 4;
    int node = te + (lane & 15);
    bool valid = node < N;
    int cn = valid ? node : N - 1;

    float c  = cnt[cn];
    float ic = 1.f / fmaxf(c, 1.f);
    const float* pr = seg32 + (size_t)cn * 32;
    v8h lo, hi;
#pragma unroll
    for (int i = 0; i < 8; ++i) {
      lo[i] = (h16)(pr[kb + i] * ic);
      hi[i] = (h16)(pr[kb + 16 + i] * ic);
    }
    v16h bf = combine16(lo, hi);

    v8f d0 = wmma16(wa0a, bf, b0a);
    v8f d1 = wmma16(wa0b, bf, b0b);
    bf = combine16(relu8h(d0), relu8h(d1));
    d0 = wmma16(wa1a, bf, b1a);
    d1 = wmma16(wa1b, bf, b1b);
    bf = combine16(relu8h(d0), relu8h(d1));
    v8f dz = wmma16(wa2f, bf, zc);   // rows 0..3 = final outputs, no relu

    if (valid && lane < 16) {
#pragma unroll
      for (int o = 0; o < 4; ++o) out[(size_t)node * 4 + o] = dz[o] + b2v[o];
    }
  }
}

extern "C" void kernel_launch(void* const* d_in, const int* in_sizes, int n_in,
                              void* d_out, int out_size, void* d_ws, size_t ws_size,
                              hipStream_t stream) {
  const float* x    = (const float*)d_in[0];
  const int*   ei   = (const int*)d_in[1];
  const float* bn_g = (const float*)d_in[2];
  const float* bn_b = (const float*)d_in[3];
  const float *ew0 = (const float*)d_in[4],  *eb0 = (const float*)d_in[5];
  const float *ew1 = (const float*)d_in[6],  *eb1 = (const float*)d_in[7];
  const float *ew2 = (const float*)d_in[8],  *eb2 = (const float*)d_in[9];
  const float *nw0 = (const float*)d_in[10], *nb0 = (const float*)d_in[11];
  const float *nw1 = (const float*)d_in[12], *nb1 = (const float*)d_in[13];
  const float *nw2 = (const float*)d_in[14], *nb2 = (const float*)d_in[15];
  const float *dw0 = (const float*)d_in[16], *db0 = (const float*)d_in[17];
  const float *dw1 = (const float*)d_in[18], *db1 = (const float*)d_in[19];
  const float *dw2 = (const float*)d_in[20], *db2 = (const float*)d_in[21];
  const float *uw0 = (const float*)d_in[22], *ub0 = (const float*)d_in[23];
  const float *uw1 = (const float*)d_in[24], *ub1 = (const float*)d_in[25];
  const float *uw2 = (const float*)d_in[26], *ub2 = (const float*)d_in[27];

  int N = in_sizes[0] / 4;
  int E = in_sizes[1] / 2;

  auto al = [](size_t o) { return (o + 63) & ~(size_t)63; };
  size_t o = 0;
  size_t off_bn    = o; o = al(o + 8 * sizeof(float));
  size_t off_cnt   = o; o = al(o + (size_t)N * sizeof(float));
  size_t off_h     = o; o = al(o + (size_t)N * 32 * sizeof(h16));
  size_t off_seg2  = o; o = al(o + (size_t)N * 2 * sizeof(float));
  size_t off_hid   = o; o = al(o + (size_t)N * 2 * sizeof(float));
  size_t off_seg32 = o; o = al(o + (size_t)N * 32 * sizeof(float));
  size_t total = o;

  char*  ws    = (char*)d_ws;
  float* bnws  = (float*)(ws + off_bn);
  float* cnt   = (float*)(ws + off_cnt);
  h16*   hbuf  = (h16*)(ws + off_h);
  float* seg2  = (float*)(ws + off_seg2);
  float* hid   = (float*)(ws + off_hid);
  float* seg32 = (float*)(ws + off_seg32);

  hipMemsetAsync(d_ws, 0, total, stream);

  k_bnstats<<<256, 256, 0, stream>>>(x, N, bnws);
  k_count<<<1024, 256, 0, stream>>>(ei, E, cnt);

  int ntilesN = (N + 15) / 16;
  int blkN = (ntilesN + WPB - 1) / WPB;
  if (blkN > 4096) blkN = 4096;
  k_embed<<<blkN, TPB, 0, stream>>>(x, N, bnws, bn_g, bn_b, ew0, eb0, ew1, eb1,
                                    ew2, eb2, hbuf);

  int ntilesE = (E + 15) / 16;
  int npairs = (ntilesE + 1) / 2;
  int blkE = (npairs + WPB - 1) / WPB;
  if (blkE > 2048) blkE = 2048;
  k_conv1<<<blkE, TPB, 0, stream>>>(ei, E, hbuf, nw0, nb0, nw1, nb1, nw2, nb2, seg2);

  int nm = 2 * N;
  k_mean2<<<(nm + 255) / 256, 256, 0, stream>>>(seg2, cnt, N, hid);

  k_conv2<<<blkE, TPB, 0, stream>>>(ei, E, hid, dw0, db0, dw1, db1, dw2, db2, seg32);

  k_deembed<<<blkN, TPB, 0, stream>>>(seg32, cnt, N, uw0, ub0, uw1, ub1, uw2, ub2,
                                      (float*)d_out);
}
